// ReferenceFFN_81406810129214
// MI455X (gfx1250) — matrix operand order repaired and checked by reading, not compiled, not run on hardware
//
#include <hip/hip_runtime.h>
#include <hip/hip_bf16.h>

// ---------------------------------------------------------------------------
// Masked SwiGLU FFN for MI455X (gfx1250).
//   G = x@Wg, U = x@Wu, Z = silu(G)*U masked to topk columns, out = Z@Wd
// Dense bf16 WMMA; B tiles staged into LDS with the CDNA5 async load-to-LDS
// path (ASYNCcnt), double-buffered, branch-free peeled pipeline.  A sched
// fence keeps all 16 fragment ds_loads in flight before the WMMA burst.
// ---------------------------------------------------------------------------

typedef __bf16 bf16_t;
typedef __bf16 v16bf __attribute__((ext_vector_type(16)));
typedef __bf16 v8bf  __attribute__((ext_vector_type(8)));
typedef float  v8f   __attribute__((ext_vector_type(8)));

#define D_MODEL 2048
#define D_FFN   8192
#define N_TOK   8192
#define TOPK    256
#define MASK_WORDS (D_FFN / 32)   // 256 u32 words per token

#define ROW_STRIDE 80             // 64B data + 16B pad: conflict-free, 16B aligned
#define MAT_BYTES  (64 * ROW_STRIDE)

// -------------------------------------------------------- async LDS helpers
__device__ __forceinline__ void async_b128(unsigned ldsOff, const void* g) {
  unsigned long long ga = (unsigned long long)g;
  asm volatile("global_load_async_to_lds_b128 %0, %1, off"
               :: "v"(ldsOff), "v"(ga) : "memory");
}
__device__ __forceinline__ void wait_async0() {
  asm volatile("s_wait_asynccnt 0" ::: "memory");
}
__device__ __forceinline__ unsigned lds_base() {
#if __has_builtin(__builtin_amdgcn_groupstaticsize)
  return __builtin_amdgcn_groupstaticsize();
#else
  return 0u;
#endif
}
__device__ __forceinline__ void sched_fence() {
#if __has_builtin(__builtin_amdgcn_sched_barrier)
  __builtin_amdgcn_sched_barrier(0);   // no instruction may cross
#endif
}

// ---------------------------------------------------------------- converts
__global__ void convert_f32_bf16(const float* __restrict__ src,
                                 bf16_t* __restrict__ dst, int n) {
  int i = blockIdx.x * blockDim.x + threadIdx.x;
  if (i < n) dst[i] = (bf16_t)src[i];
}

// dst[cols][rows] (bf16) = transpose of src[rows][cols] (f32); dims % 32 == 0
__global__ void transpose_convert(const float* __restrict__ src,
                                  bf16_t* __restrict__ dst,
                                  int rows, int cols) {
  __shared__ float tile[32][33];
  int c0 = blockIdx.x * 32, r0 = blockIdx.y * 32;
  int tx = threadIdx.x, ty = threadIdx.y;           // block (32, 8)
#pragma unroll
  for (int i = 0; i < 32; i += 8)
    tile[ty + i][tx] = src[(long)(r0 + ty + i) * cols + (c0 + tx)];
  __syncthreads();
#pragma unroll
  for (int i = 0; i < 32; i += 8)
    dst[(long)(c0 + ty + i) * rows + (r0 + tx)] = (bf16_t)tile[tx][ty + i];
}

// ---------------------------------------------------------------- mask bits
__global__ void zero_u32(unsigned* __restrict__ p, int n) {
  int i = blockIdx.x * blockDim.x + threadIdx.x;
  if (i < n) p[i] = 0u;
}

__global__ void scatter_mask(const int* __restrict__ topk,
                             unsigned* __restrict__ mask) {
  int i = blockIdx.x * blockDim.x + threadIdx.x;    // over N_TOK*TOPK
  if (i >= N_TOK * TOPK) return;
  int tok = i / TOPK;
  int f = topk[i];
  atomicOr(&mask[tok * MASK_WORDS + (f >> 5)], 1u << (f & 31));
}

// ---------------------------------------------------------------- fragments
// A (16x32 bf16, MxK): lanes 0-15 -> M=lane, K 0..15;
//                      lanes 16-31 -> M=lane-16, K 16..31
__device__ __forceinline__ v16bf load_frag_a(const bf16_t* __restrict__ base,
                                             int ld, int m0, int k0, int lane) {
  int m = lane & 15, kh = (lane >> 4) & 1;
  const bf16_t* p = base + (long)(m0 + m) * ld + k0 + kh * 16;
  v8bf lo = *(const v8bf*)p;
  v8bf hi = *(const v8bf*)(p + 8);
  return __builtin_shufflevector(lo, hi, 0, 1, 2, 3, 4, 5, 6, 7,
                                         8, 9, 10, 11, 12, 13, 14, 15);
}

// B (32x16 bf16, KxN) fragment from an LDS-staged tile row (padded rows):
// lanes 0-15 -> N=lane, K 0..15; lanes 16-31 -> N=lane-16, K 16..31
__device__ __forceinline__ v16bf lds_frag(const char* p) {
  v8bf lo = *(const v8bf*)p;
  v8bf hi = *(const v8bf*)(p + 16);
  return __builtin_shufflevector(lo, hi, 0, 1, 2, 3, 4, 5, 6, 7,
                                         8, 9, 10, 11, 12, 13, 14, 15);
}

// ------------------------------------------------------- WMMA burst helpers
// All fragment ds_loads are issued first; the sched fence stops the scheduler
// from sinking them back to their consumers, so the waits become staged and
// the 8 WMMAs issue back-to-back while later loads are still in flight.
__device__ __forceinline__ void p1_step(const char* bufp, const v16bf& a,
                                        unsigned rbase, v8f* accg, v8f* accu) {
  v16bf bg[4], bu[4];
#pragma unroll
  for (int s = 0; s < 4; ++s) {
    unsigned row = rbase + (unsigned)(s * 16) * ROW_STRIDE;
    bg[s] = lds_frag(bufp + row);
    bu[s] = lds_frag(bufp + MAT_BYTES + row);
  }
  sched_fence();
#pragma unroll
  for (int s = 0; s < 4; ++s) {
    accg[s] = __builtin_amdgcn_wmma_f32_16x16x32_bf16(
        false, a, false, bg[s], (short)0, accg[s], false, false);
    accu[s] = __builtin_amdgcn_wmma_f32_16x16x32_bf16(
        false, a, false, bu[s], (short)0, accu[s], false, false);
  }
}

__device__ __forceinline__ void p2_step(const char* bufp, const v16bf& a,
                                        unsigned rbase, v8f* acc) {
  v16bf b[4];
#pragma unroll
  for (int s = 0; s < 4; ++s)
    b[s] = lds_frag(bufp + rbase + (unsigned)(s * 16) * ROW_STRIDE);
  sched_fence();
#pragma unroll
  for (int s = 0; s < 4; ++s)
    acc[s] = __builtin_amdgcn_wmma_f32_16x16x32_bf16(
        false, a, false, b[s], (short)0, acc[s], false, false);
}

// ---------------------------------------------------------------- phase 1
// Fused gate+up GEMM + SwiGLU + topk mask -> Z (bf16).
// Block = 256 threads (8 waves); tile M=128 (16/wave), N=64.
__global__ void __launch_bounds__(256) gate_up_swiglu(
    const bf16_t* __restrict__ x16,    // [N_TOK][D_MODEL]
    const bf16_t* __restrict__ wgT,    // [D_FFN][D_MODEL]
    const bf16_t* __restrict__ wuT,    // [D_FFN][D_MODEL]
    const unsigned* __restrict__ mask, // [N_TOK][MASK_WORDS]
    bf16_t* __restrict__ Z) {          // [N_TOK][D_FFN]
  extern __shared__ char smem[];                 // 2 * 2 * MAT_BYTES
  const unsigned BUF_BYTES = 2 * MAT_BYTES;

  int lane = threadIdx.x & 31;
  int wave = threadIdx.x >> 5;
  int m0 = blockIdx.x * 128 + wave * 16;
  int n0 = blockIdx.y * 64;

  // per-thread staging assignment: 512 x 16B chunks, 2 per thread (32B)
  int t = threadIdx.x;
  int mMat = t >> 7;                         // 0 = gate, 1 = up
  int rRow = (t & 127) >> 1;                 // 0..63
  int qOff = (t & 1) * 32;                   // byte offset in row: 0 or 32
  const char* gRow =
      (const char*)((mMat ? wuT : wgT) + (long)(n0 + rRow) * D_MODEL) + qOff;
  unsigned ldsDst0 = lds_base() + mMat * MAT_BYTES + rRow * ROW_STRIDE + qOff;
  unsigned ldsDst1 = ldsDst0 + BUF_BYTES;

  v8f zero = {0.f, 0.f, 0.f, 0.f, 0.f, 0.f, 0.f, 0.f};
  v8f accg[4] = {zero, zero, zero, zero};
  v8f accu[4] = {zero, zero, zero, zero};

  int nfrag = lane & 15;
  unsigned rbase = (unsigned)nfrag * ROW_STRIDE + ((lane >> 4) & 1) * 32;
  const char* buf0 = smem;
  const char* buf1 = smem + BUF_BYTES;

  // prologue: fill buffer 0 with k0 = 0
  async_b128(ldsDst0, gRow);
  async_b128(ldsDst0 + 16, gRow + 16);

  for (int k0 = 0; k0 < D_MODEL - 64; k0 += 64) {
    v16bf a0 = load_frag_a(x16, D_MODEL, m0, k0, lane);      // overlaps barrier
    wait_async0();
    __syncthreads();
    { const char* g = gRow + 2 * (k0 + 32);
      async_b128(ldsDst1, g); async_b128(ldsDst1 + 16, g + 16); }
    p1_step(buf0, a0, rbase, accg, accu);

    v16bf a1 = load_frag_a(x16, D_MODEL, m0, k0 + 32, lane);
    wait_async0();
    __syncthreads();
    { const char* g = gRow + 2 * (k0 + 64);
      async_b128(ldsDst0, g); async_b128(ldsDst0 + 16, g + 16); }
    p1_step(buf1, a1, rbase, accg, accu);
  }
  // tail pair (k0 = D_MODEL-64): prefetch last chunk, no prefetch after it
  {
    v16bf a0 = load_frag_a(x16, D_MODEL, m0, D_MODEL - 64, lane);
    wait_async0();
    __syncthreads();
    { const char* g = gRow + 2 * (D_MODEL - 32);
      async_b128(ldsDst1, g); async_b128(ldsDst1 + 16, g + 16); }
    p1_step(buf0, a0, rbase, accg, accu);

    v16bf a1 = load_frag_a(x16, D_MODEL, m0, D_MODEL - 32, lane);
    wait_async0();
    __syncthreads();
    p1_step(buf1, a1, rbase, accg, accu);
  }

  // C/D layout: lane<16 -> rows r, lane>=16 -> rows r+8; col = lane&15
  int row_off = (lane >> 4) ? 8 : 0;
#pragma unroll
  for (int s = 0; s < 4; ++s) {
    int col = n0 + s * 16 + nfrag;
#pragma unroll
    for (int r = 0; r < 8; ++r) {
      int row = m0 + r + row_off;
      float g = accg[s][r];
      float u = accu[s][r];
      float z = (g / (1.0f + __expf(-g))) * u;   // silu(g) * u
      unsigned mw = mask[row * MASK_WORDS + (col >> 5)];
      z = ((mw >> (col & 31)) & 1u) ? z : 0.0f;
      Z[(long)row * D_FFN + col] = (bf16_t)z;
    }
  }
}

// ---------------------------------------------------------------- phase 2
// out = Z @ Wd  (A = Z bf16 row-major, B from wdT[D_MODEL][D_FFN]).
__global__ void __launch_bounds__(256) down_proj(
    const bf16_t* __restrict__ Z,     // [N_TOK][D_FFN]
    const bf16_t* __restrict__ wdT,   // [D_MODEL][D_FFN]
    float* __restrict__ out) {        // [N_TOK][D_MODEL]
  extern __shared__ char smem[];                 // 2 * MAT_BYTES
  const unsigned BUF_BYTES = MAT_BYTES;

  int lane = threadIdx.x & 31;
  int wave = threadIdx.x >> 5;
  int m0 = blockIdx.x * 128 + wave * 16;
  int n0 = blockIdx.y * 64;

  // staging: 256 x 16B chunks, 1 per thread
  int t = threadIdx.x;
  int rRow = t >> 2;                          // 0..63
  int qOff = (t & 3) * 16;                    // 0,16,32,48
  const char* gRow = (const char*)(wdT + (long)(n0 + rRow) * D_FFN) + qOff;
  unsigned ldsDst0 = lds_base() + rRow * ROW_STRIDE + qOff;
  unsigned ldsDst1 = ldsDst0 + BUF_BYTES;

  v8f zero = {0.f, 0.f, 0.f, 0.f, 0.f, 0.f, 0.f, 0.f};
  v8f acc[4] = {zero, zero, zero, zero};

  int nfrag = lane & 15;
  unsigned rbase = (unsigned)nfrag * ROW_STRIDE + ((lane >> 4) & 1) * 32;
  const char* buf0 = smem;
  const char* buf1 = smem + BUF_BYTES;

  async_b128(ldsDst0, gRow);                  // prologue, buf 0 (k0 = 0)

  for (int k0 = 0; k0 < D_FFN - 64; k0 += 64) {
    v16bf a0 = load_frag_a(Z, D_FFN, m0, k0, lane);
    wait_async0();
    __syncthreads();
    async_b128(ldsDst1, gRow + 2 * (k0 + 32));
    p2_step(buf0, a0, rbase, acc);

    v16bf a1 = load_frag_a(Z, D_FFN, m0, k0 + 32, lane);
    wait_async0();
    __syncthreads();
    async_b128(ldsDst0, gRow + 2 * (k0 + 64));
    p2_step(buf1, a1, rbase, acc);
  }
  {
    v16bf a0 = load_frag_a(Z, D_FFN, m0, D_FFN - 64, lane);
    wait_async0();
    __syncthreads();
    async_b128(ldsDst1, gRow + 2 * (D_FFN - 32));
    p2_step(buf0, a0, rbase, acc);

    v16bf a1 = load_frag_a(Z, D_FFN, m0, D_FFN - 32, lane);
    wait_async0();
    __syncthreads();
    p2_step(buf1, a1, rbase, acc);
  }

  int row_off = (lane >> 4) ? 8 : 0;
#pragma unroll
  for (int s = 0; s < 4; ++s) {
    int col = n0 + s * 16 + nfrag;
#pragma unroll
    for (int r = 0; r < 8; ++r) {
      int row = m0 + r + row_off;
      out[(long)row * D_MODEL + col] = acc[s][r];
    }
  }
}

// ---------------------------------------------------------------- launch
extern "C" void kernel_launch(void* const* d_in, const int* in_sizes, int n_in,
                              void* d_out, int out_size, void* d_ws,
                              size_t ws_size, hipStream_t stream) {
  const float* x      = (const float*)d_in[0];   // [N_TOK][D_MODEL]
  const int*   topk   = (const int*)d_in[1];     // [N_TOK][TOPK]
  const float* w_gate = (const float*)d_in[2];   // [D_MODEL][D_FFN]
  const float* w_up   = (const float*)d_in[3];   // [D_MODEL][D_FFN]
  const float* w_down = (const float*)d_in[4];   // [D_FFN][D_MODEL]
  float* out = (float*)d_out;

  // Workspace carve-out (~264 MB total)
  char* ws = (char*)d_ws;
  size_t off = 0;
  bf16_t* x16 = (bf16_t*)(ws + off); off += (size_t)N_TOK * D_MODEL * 2;
  bf16_t* wgT = (bf16_t*)(ws + off); off += (size_t)D_FFN * D_MODEL * 2;
  bf16_t* wuT = (bf16_t*)(ws + off); off += (size_t)D_FFN * D_MODEL * 2;
  bf16_t* wdT = (bf16_t*)(ws + off); off += (size_t)D_MODEL * D_FFN * 2;
  bf16_t* Zb  = (bf16_t*)(ws + off); off += (size_t)N_TOK * D_FFN * 2;
  unsigned* maskbits = (unsigned*)(ws + off); off += (size_t)N_TOK * MASK_WORDS * 4;
  (void)ws_size; (void)in_sizes; (void)n_in; (void)out_size;

  // 1) precision conversion + weight transposition (K-contiguous bf16)
  {
    int n = N_TOK * D_MODEL;
    convert_f32_bf16<<<(n + 255) / 256, 256, 0, stream>>>(x, x16, n);
  }
  transpose_convert<<<dim3(D_FFN / 32, D_MODEL / 32), dim3(32, 8), 0, stream>>>(
      w_gate, wgT, D_MODEL, D_FFN);
  transpose_convert<<<dim3(D_FFN / 32, D_MODEL / 32), dim3(32, 8), 0, stream>>>(
      w_up, wuT, D_MODEL, D_FFN);
  transpose_convert<<<dim3(D_MODEL / 32, D_FFN / 32), dim3(32, 8), 0, stream>>>(
      w_down, wdT, D_FFN, D_MODEL);

  // 2) top-k mask bitfield
  {
    int nw = N_TOK * MASK_WORDS;
    zero_u32<<<(nw + 255) / 256, 256, 0, stream>>>(maskbits, nw);
    int np = N_TOK * TOPK;
    scatter_mask<<<(np + 255) / 256, 256, 0, stream>>>(topk, maskbits);
  }

  // 3) fused gate/up GEMM + SwiGLU + mask -> Z   (LDS: 2 bufs * 2 mats)
  gate_up_swiglu<<<dim3(N_TOK / 128, D_FFN / 64), 256,
                   2 * 2 * MAT_BYTES, stream>>>(x16, wgT, wuT, maskbits, Zb);

  // 4) down projection GEMM -> out (fp32)       (LDS: 2 bufs * 1 mat)
  down_proj<<<dim3(N_TOK / 128, D_MODEL / 64), 256,
              2 * MAT_BYTES, stream>>>(Zb, wdT, out);
}
